// simple_GraphTransformer_regression_14912126452014
// MI455X (gfx1250) — matrix operand order repaired and checked by reading, not compiled, not run on hardware
//
#include <hip/hip_runtime.h>
#include <math.h>

#define NN 50000
#define NE 800000
#define HD 128
#define NH 8
#define DH 16
#define NG 64

typedef __attribute__((ext_vector_type(16))) _Float16 v16h;
typedef __attribute__((ext_vector_type(8)))  _Float16 v8h;
typedef __attribute__((ext_vector_type(8)))  float    v8f;

// ---------------- utility ----------------
__global__ void k_init_f32(float* __restrict__ p, float v, int n) {
  int i = blockIdx.x * blockDim.x + threadIdx.x;
  if (i < n) p[i] = v;
}

__device__ __forceinline__ void atomicMaxF(float* a, float v) {
  if (v >= 0.0f) atomicMax((int*)a, __float_as_int(v));
  else           atomicMin((unsigned int*)a, __float_as_uint(v));
}

// ---------------- layer 1 projections (in_dim = 1) ----------------
__global__ void k_proj1(const float* __restrict__ X,
                        const float* __restrict__ Wq, const float* __restrict__ bq,
                        const float* __restrict__ Wk, const float* __restrict__ bk,
                        const float* __restrict__ Wv, const float* __restrict__ bv,
                        const float* __restrict__ Ws, const float* __restrict__ bs,
                        float* __restrict__ q, float* __restrict__ k,
                        float* __restrict__ v, float* __restrict__ h) {
  int idx = blockIdx.x * blockDim.x + threadIdx.x;
  if (idx >= NN * HD) return;
  int n = idx >> 7, c = idx & 127;
  float xv = X[n];
  q[idx] = xv * Wq[c] + bq[c];
  k[idx] = xv * Wk[c] + bk[c];
  v[idx] = xv * Wv[c] + bv[c];
  h[idx] = xv * Ws[c] + bs[c];
}

// ---------------- fp16 conversion ----------------
__global__ void k_cvt_x(const float* __restrict__ x, _Float16* __restrict__ xh, int n) {
  int i = blockIdx.x * blockDim.x + threadIdx.x;
  if (i < n) xh[i] = (_Float16)x[i];
}

// WT[n*128 + k] = (f16) W[k*128 + n]   (column-major copy for B fragments)
__global__ void k_cvt_wT(const float* __restrict__ W, _Float16* __restrict__ WT) {
  int idx = blockIdx.x * blockDim.x + threadIdx.x;
  if (idx >= HD * HD) return;
  int n = idx >> 7, k = idx & 127;
  WT[idx] = (_Float16)W[k * HD + n];
}

// ------- fused WMMA GEMM: {Yq,Yk,Yv,Ys}(N x 128) = Xh(N x 128) @ {Wq,Wk,Wv,Ws} + b -------
// block = 256 threads (8 waves); block computes 16 rows x 128 cols for 4 outputs.
// wave w -> column tile [16w, 16w+16). K = 128 = 4 x 32.
// A fragment is loaded ONCE per K-step and fed to 4 independent WMMA chains.
__global__ __launch_bounds__(256) void k_gemm_wmma4(
    const _Float16* __restrict__ Xh,
    const _Float16* __restrict__ WTq, const _Float16* __restrict__ WTk,
    const _Float16* __restrict__ WTv, const _Float16* __restrict__ WTs,
    const float* __restrict__ bq, const float* __restrict__ bk,
    const float* __restrict__ bv, const float* __restrict__ bs,
    float* __restrict__ Yq, float* __restrict__ Yk,
    float* __restrict__ Yv, float* __restrict__ Ys) {
  const int lane = threadIdx.x & 31;
  const int wv   = threadIdx.x >> 5;
  const int row0 = blockIdx.x << 4;
  const int col0 = wv << 4;
  const int hs   = lane >> 4;   // half-wave select
  const int l16  = lane & 15;
  const size_t wo = (size_t)(col0 + l16) * HD;
  const _Float16* __restrict__ xrow = Xh + (size_t)(row0 + l16) * HD;
  const int ko = hs * 16;
  v8f aq = {}, ak = {}, av = {}, as = {};
#pragma unroll
  for (int ks = 0; ks < 4; ++ks) {
    const int kb = ks * 32;
    v16h a, b0, b1, b2, b3;
    // A 16x32 f16 layout: lanes 0-15 K=kb..kb+15, lanes 16-31 K=kb+16..kb+31
    ((v8h*)&a)[0] = *(const v8h*)(xrow + kb + ko);
    ((v8h*)&a)[1] = *(const v8h*)(xrow + kb + 8 + ko);
    // B 32x16 f16 layout: lanes 0-15 K=kb..kb+15, lanes 16-31 K=kb+16..kb+31
    const int bo = kb + hs * 16;
    b0 = *(const v16h*)(WTq + wo + bo);
    b1 = *(const v16h*)(WTk + wo + bo);
    b2 = *(const v16h*)(WTv + wo + bo);
    b3 = *(const v16h*)(WTs + wo + bo);
    aq = __builtin_amdgcn_wmma_f32_16x16x32_f16(false, a, false, b0, (short)0, aq, false, false);
    ak = __builtin_amdgcn_wmma_f32_16x16x32_f16(false, a, false, b1, (short)0, ak, false, false);
    av = __builtin_amdgcn_wmma_f32_16x16x32_f16(false, a, false, b2, (short)0, av, false, false);
    as = __builtin_amdgcn_wmma_f32_16x16x32_f16(false, a, false, b3, (short)0, as, false, false);
  }
  const float bbq = bq[col0 + l16], bbk = bk[col0 + l16];
  const float bbv = bv[col0 + l16], bbs = bs[col0 + l16];
#pragma unroll
  for (int r = 0; r < 8; ++r) {
    const size_t o = (size_t)(row0 + r + hs * 8) * HD + col0 + l16;
    Yq[o] = aq[r] + bbq;
    Yk[o] = ak[r] + bbk;
    Yv[o] = av[r] + bbv;
    Ys[o] = as[r] + bbs;
  }
}

// ---------------- edge attention ----------------
__global__ void k_attn_logits(const float* __restrict__ q, const float* __restrict__ k,
                              const int* __restrict__ src, const int* __restrict__ dst,
                              float* __restrict__ alpha, float* __restrict__ amax) {
  int idx = blockIdx.x * blockDim.x + threadIdx.x;
  if (idx >= NE * NH) return;
  int e = idx >> 3, h = idx & 7;
  int s = src[e], d = dst[e];
  const float* qp = q + (size_t)d * HD + h * DH;
  const float* kp = k + (size_t)s * HD + h * DH;
  float acc = 0.f;
#pragma unroll
  for (int j = 0; j < DH; ++j) acc += qp[j] * kp[j];
  acc *= 0.25f;  // 1/sqrt(16)
  alpha[idx] = acc;
  atomicMaxF(&amax[d * NH + h], acc);
}

__global__ void k_amax_clean(float* __restrict__ amax, int n) {
  int i = blockIdx.x * blockDim.x + threadIdx.x;
  if (i >= n) return;
  float a = amax[i];
  amax[i] = __builtin_isfinite(a) ? a : 0.0f;
}

__global__ void k_attn_exp(const int* __restrict__ dst, const float* __restrict__ amax,
                           float* __restrict__ alpha, float* __restrict__ denom) {
  int idx = blockIdx.x * blockDim.x + threadIdx.x;
  if (idx >= NE * NH) return;
  int e = idx >> 3, h = idx & 7;
  int d = dst[e];
  float ex = expf(alpha[idx] - amax[d * NH + h]);
  alpha[idx] = ex;
  atomicAdd(&denom[d * NH + h], ex);
}

// normalize once per (edge, head): w = ex / (denom[dst] + eps), in place over ex
__global__ void k_attn_wnorm(const int* __restrict__ dst, const float* __restrict__ denom,
                             float* __restrict__ wgt) {
  int idx = blockIdx.x * blockDim.x + threadIdx.x;
  if (idx >= NE * NH) return;
  int e = idx >> 3, h = idx & 7;
  wgt[idx] = wgt[idx] / (denom[dst[e] * NH + h] + 1e-16f);
}

// scatter: out[dst] += v[src] * w ; one float4 per thread
__global__ void k_attn_scatter(const int* __restrict__ src, const int* __restrict__ dst,
                               const float* __restrict__ wgt,
                               const float* __restrict__ v, float* __restrict__ out) {
  int idx = blockIdx.x * blockDim.x + threadIdx.x;
  if (idx >= NE * (HD / 4)) return;
  int e = idx >> 5, c4 = (idx & 31) << 2, h = c4 >> 4;
  int s = src[e], d = dst[e];
  float w = wgt[e * NH + h];
  const float4 vv = *(const float4*)(v + (size_t)s * HD + c4);
  float* op = out + (size_t)d * HD + c4;
  atomicAdd(op + 0, vv.x * w);
  atomicAdd(op + 1, vv.y * w);
  atomicAdd(op + 2, vv.z * w);
  atomicAdd(op + 3, vv.w * w);
}

// ---------------- batchnorm on relu(h) ----------------
__global__ __launch_bounds__(256) void k_bn_stats(const float* __restrict__ h,
                                                  float* __restrict__ gsum,
                                                  float* __restrict__ gsumsq) {
  int c = threadIdx.x & 127;
  int slice = blockIdx.x * 2 + (threadIdx.x >> 7);   // 0..255 (grid 128)
  float s = 0.f, ss = 0.f;
  for (int n = slice; n < NN; n += 256) {
    float x = h[(size_t)n * HD + c];
    x = x > 0.f ? x : 0.f;
    s += x; ss += x * x;
  }
  atomicAdd(&gsum[c], s);
  atomicAdd(&gsumsq[c], ss);
}

__global__ void k_bn_finalize(float* __restrict__ gsum, float* __restrict__ gsumsq) {
  int c = threadIdx.x;
  if (c >= HD) return;
  float mu  = gsum[c] * (1.0f / NN);
  float var = gsumsq[c] * (1.0f / NN) - mu * mu;
  gsum[c]   = mu;                       // reuse as mean
  gsumsq[c] = rsqrtf(var + 1e-5f);      // reuse as invstd
}

__global__ void k_bn_apply(const float* __restrict__ h, const float* __restrict__ mu,
                           const float* __restrict__ is, const float* __restrict__ gamma,
                           const float* __restrict__ beta, float* __restrict__ x,
                           int residual) {
  int idx = blockIdx.x * blockDim.x + threadIdx.x;
  if (idx >= NN * HD) return;
  int c = idx & 127;
  float v = h[idx]; v = v > 0.f ? v : 0.f;
  float y = (v - mu[c]) * is[c] * gamma[c] + beta[c];
  if (residual) y += x[idx];
  x[idx] = y;
}

// ---------------- pooling + head ----------------
__global__ void k_pool_sum(const float* __restrict__ x, const int* __restrict__ batch,
                           float* __restrict__ psum) {
  int idx = blockIdx.x * blockDim.x + threadIdx.x;
  if (idx >= NN * HD) return;
  int n = idx >> 7, c = idx & 127;
  atomicAdd(&psum[batch[n] * HD + c], x[idx]);
}

__global__ void k_pool_cnt(const int* __restrict__ batch, float* __restrict__ cnt) {
  int i = blockIdx.x * blockDim.x + threadIdx.x;
  if (i < NN) atomicAdd(&cnt[batch[i]], 1.0f);
}

__global__ __launch_bounds__(128) void k_head(const float* __restrict__ psum,
                                              const float* __restrict__ cnt,
                                              const float* __restrict__ Wr,
                                              const float* __restrict__ br,
                                              float* __restrict__ out) {
  __shared__ float red[128];
  int g = blockIdx.x, c = threadIdx.x;
  float cc = cnt[g]; cc = cc > 1.f ? cc : 1.f;
  float m = psum[g * HD + c] / cc;
  out[NG + g * HD + c] = m;      // x_mean
  red[c] = m * Wr[c];
  __syncthreads();
  for (int s = 64; s > 0; s >>= 1) {
    if (c < s) red[c] += red[c + s];
    __syncthreads();
  }
  if (c == 0) out[g] = red[0] + br[0];   // pred
}

// ---------------- host orchestration ----------------
extern "C" void kernel_launch(void* const* d_in, const int* in_sizes, int n_in,
                              void* d_out, int out_size, void* d_ws, size_t ws_size,
                              hipStream_t stream) {
  (void)in_sizes; (void)n_in; (void)out_size; (void)ws_size;
  const float* X     = (const float*)d_in[0];
  const int*   ei    = (const int*)d_in[1];
  const int*   batch = (const int*)d_in[2];
  const float* Wq1 = (const float*)d_in[3];  const float* bq1 = (const float*)d_in[4];
  const float* Wk1 = (const float*)d_in[5];  const float* bk1 = (const float*)d_in[6];
  const float* Wv1 = (const float*)d_in[7];  const float* bv1 = (const float*)d_in[8];
  const float* Ws1 = (const float*)d_in[9];  const float* bs1 = (const float*)d_in[10];
  const float* Wq  = (const float*)d_in[11]; const float* bq  = (const float*)d_in[12];
  const float* Wk  = (const float*)d_in[13]; const float* bk  = (const float*)d_in[14];
  const float* Wv  = (const float*)d_in[15]; const float* bv  = (const float*)d_in[16];
  const float* Ws  = (const float*)d_in[17]; const float* bs  = (const float*)d_in[18];
  const float* bng = (const float*)d_in[19]; const float* bnb = (const float*)d_in[20];
  const float* Wr  = (const float*)d_in[21]; const float* br  = (const float*)d_in[22];
  const int* src = ei;
  const int* dst = ei + NE;
  float* out = (float*)d_out;

  // --- workspace carve ---
  char*  ws  = (char*)d_ws;
  size_t off = 0;
  auto bump = [&](size_t bytes) -> void* {
    void* p = ws + off;
    off = (off + bytes + 255) & ~(size_t)255;
    return p;
  };
  const size_t NF = (size_t)NN * HD;
  float*    xb    = (float*)bump(NF * 4);
  float*    hb    = (float*)bump(NF * 4);
  float*    qb    = (float*)bump(NF * 4);
  float*    kbuf  = (float*)bump(NF * 4);
  float*    vbuf  = (float*)bump(NF * 4);
  float*    alpha = (float*)bump((size_t)NE * NH * 4);
  float*    amax  = (float*)bump((size_t)NN * NH * 4);
  float*    denom = (float*)bump((size_t)NN * NH * 4);
  _Float16* xh    = (_Float16*)bump(NF * 2);
  _Float16* wT[4];
  for (int i = 0; i < 4; ++i) wT[i] = (_Float16*)bump((size_t)HD * HD * 2);
  float*    gsum  = (float*)bump(2 * HD * 4);    // [mean|invstd] after finalize
  float*    gsq   = gsum + HD;
  float*    psum  = (float*)bump((NG * HD + NG) * 4);
  float*    pcnt  = psum + NG * HD;

  const dim3 B(256);
  auto G = [](long long n) { return dim3((unsigned)((n + 255) / 256)); };
  const long long nNF = (long long)NN * HD;
  const long long nEH = (long long)NE * NH;
  const long long nE4 = (long long)NE * (HD / 4);

  auto run_attention = [&]() {
    k_init_f32<<<G(NN * NH), B, 0, stream>>>(amax, -INFINITY, NN * NH);
    k_init_f32<<<G(NN * NH), B, 0, stream>>>(denom, 0.0f, NN * NH);
    k_attn_logits<<<G(nEH), B, 0, stream>>>(qb, kbuf, src, dst, alpha, amax);
    k_amax_clean<<<G(NN * NH), B, 0, stream>>>(amax, NN * NH);
    k_attn_exp<<<G(nEH), B, 0, stream>>>(dst, amax, alpha, denom);
    k_attn_wnorm<<<G(nEH), B, 0, stream>>>(dst, denom, alpha);
    k_attn_scatter<<<G(nE4), B, 0, stream>>>(src, dst, alpha, vbuf, hb);
  };
  auto run_bn = [&](int lay, int residual) {
    k_init_f32<<<1, 256, 0, stream>>>(gsum, 0.0f, 2 * HD);
    k_bn_stats<<<128, 256, 0, stream>>>(hb, gsum, gsq);
    k_bn_finalize<<<1, 128, 0, stream>>>(gsum, gsq);
    k_bn_apply<<<G(nNF), B, 0, stream>>>(hb, gsum, gsq, bng + lay * HD,
                                         bnb + lay * HD, xb, residual);
  };

  // ----- layer 1 (in_dim = 1, scalar projections) -----
  k_proj1<<<G(nNF), B, 0, stream>>>(X, Wq1, bq1, Wk1, bk1, Wv1, bv1, Ws1, bs1,
                                    qb, kbuf, vbuf, hb);
  run_attention();
  run_bn(0, /*residual=*/0);

  // ----- layers 2..5 (fused WMMA projections) -----
  for (int i = 0; i < 4; ++i) {
    k_cvt_x<<<G(nNF), B, 0, stream>>>(xb, xh, (int)nNF);
    k_cvt_wT<<<G(HD * HD), B, 0, stream>>>(Wq + (size_t)i * HD * HD, wT[0]);
    k_cvt_wT<<<G(HD * HD), B, 0, stream>>>(Wk + (size_t)i * HD * HD, wT[1]);
    k_cvt_wT<<<G(HD * HD), B, 0, stream>>>(Wv + (size_t)i * HD * HD, wT[2]);
    k_cvt_wT<<<G(HD * HD), B, 0, stream>>>(Ws + (size_t)i * HD * HD, wT[3]);
    k_gemm_wmma4<<<NN / 16, 256, 0, stream>>>(xh, wT[0], wT[1], wT[2], wT[3],
                                              bq + i * HD, bk + i * HD,
                                              bv + i * HD, bs + i * HD,
                                              qb, kbuf, vbuf, hb);
    run_attention();
    run_bn(i + 1, /*residual=*/1);
  }

  // ----- global mean pool + regression head -----
  k_init_f32<<<G(NG * HD + NG), B, 0, stream>>>(psum, 0.0f, NG * HD + NG);
  k_pool_sum<<<G(nNF), B, 0, stream>>>(xb, batch, psum);
  k_pool_cnt<<<G(NN), B, 0, stream>>>(batch, pcnt);
  k_head<<<NG, 128, 0, stream>>>(psum, pcnt, Wr, br, out);
}
